// PointnetSAModuleMSG_5334349381897
// MI455X (gfx1250) — matrix operand order repaired and checked by reading, not compiled, not run on hardware
//
#include <hip/hip_runtime.h>
#include <stdint.h>

// ---------------------------------------------------------------------------
// Types for WMMA
// ---------------------------------------------------------------------------
typedef __attribute__((ext_vector_type(16))) __bf16 v16bf;
typedef __attribute__((ext_vector_type(8)))  float  v8f;

struct U8x32 { uint4 a; uint4 b; };   // 32 bytes == v16bf

// A-operand tile (16x32 bf16) from LDS, row-major activation buffer, stride ldw halves.
// Layout (matches B): lane<16 holds row=lane, K=[0,16);
// lane>=16 holds row=lane-16, K=[16,32)  (relative to kt*32).
static __device__ __forceinline__ v16bf load_a_tile(const __bf16* buf, int ldw,
                                                    int mt, int kt, int lane) {
    int m  = mt * 16 + (lane & 15);
    int k0 = kt * 32 + ((lane < 16) ? 0 : 16);
    const uint4* p0 = reinterpret_cast<const uint4*>(buf + m * ldw + k0);
    const uint4* p1 = reinterpret_cast<const uint4*>(buf + m * ldw + k0 + 8);
    U8x32 u; u.a = *p0; u.b = *p1;
    return __builtin_bit_cast(v16bf, u);
}

// B-operand tile (32x16 bf16) from global weights, row-major (Cout x KPad).
// Lanes 0-15: K=[0,16), lanes 16-31: K=[16,32); N = lane%16 + nt*16.
static __device__ __forceinline__ v16bf load_b_tile(const __bf16* __restrict__ W,
                                                    int kpad, int nt, int kt, int lane) {
    int n  = nt * 16 + (lane & 15);
    int k0 = kt * 32 + ((lane < 16) ? 0 : 16);
    const uint4* p = reinterpret_cast<const uint4*>(W + n * kpad + k0);
    U8x32 u; u.a = p[0]; u.b = p[1];
    return __builtin_bit_cast(v16bf, u);
}

// ---------------------------------------------------------------------------
// Kernel 1: features (B,64,N) -> featT (B,N,64)
// ---------------------------------------------------------------------------
__global__ void transpose_feat_kernel(const float* __restrict__ features,
                                      float* __restrict__ featT) {
    const int N = 8192, C = 64;
    size_t t = (size_t)blockIdx.x * blockDim.x + threadIdx.x;
    if (t >= (size_t)4 * N * C) return;
    int c = (int)(t & 63);
    size_t bn = t >> 6;
    int n = (int)(bn & 8191);
    int b = (int)(bn >> 13);
    featT[t] = features[((size_t)b * C + c) * N + n];
}

// ---------------------------------------------------------------------------
// Kernel 2: furthest point sampling. One 1024-thread block per batch.
// Writes fps indices (int32) and new_xyz (B,P,3) to d_out head.
// ---------------------------------------------------------------------------
__global__ __launch_bounds__(1024)
void fps_kernel(const float* __restrict__ xyz, int* __restrict__ fpsIdx,
                float* __restrict__ newxyz) {
    const int N = 8192, P = 2048, T = 1024, PER = N / T;  // 8 points per thread
    int b = blockIdx.x;
    int tid = threadIdx.x;
    int lane = tid & 31, wave = tid >> 5;

    __shared__ float s_val[32];
    __shared__ int   s_idx[32];
    __shared__ float s_lastp[3];

    float dmin[PER];
#pragma unroll
    for (int j = 0; j < PER; ++j) dmin[j] = 1e10f;

    if (tid == 0) {
        float x0 = xyz[((size_t)b * N) * 3 + 0];
        float y0 = xyz[((size_t)b * N) * 3 + 1];
        float z0 = xyz[((size_t)b * N) * 3 + 2];
        fpsIdx[b * P + 0] = 0;
        newxyz[((size_t)b * P) * 3 + 0] = x0;
        newxyz[((size_t)b * P) * 3 + 1] = y0;
        newxyz[((size_t)b * P) * 3 + 2] = z0;
        s_lastp[0] = x0; s_lastp[1] = y0; s_lastp[2] = z0;
    }
    __syncthreads();

    for (int i = 1; i < P; ++i) {
        float lx = s_lastp[0], ly = s_lastp[1], lz = s_lastp[2];
        float best = -1.0f; int bestn = 0;
#pragma unroll
        for (int j = 0; j < PER; ++j) {
            int n = tid + j * T;
            const float* p = xyz + ((size_t)b * N + n) * 3;
            float dx = p[0] - lx, dy = p[1] - ly, dz = p[2] - lz;
            float d = dx * dx + dy * dy + dz * dz;
            dmin[j] = fminf(dmin[j], d);
            if (dmin[j] > best) { best = dmin[j]; bestn = n; }
        }
        // wave32 argmax (lowest index on tie)
#pragma unroll
        for (int off = 16; off >= 1; off >>= 1) {
            float ov = __shfl_xor(best, off, 32);
            int   on = __shfl_xor(bestn, off, 32);
            if (ov > best || (ov == best && on < bestn)) { best = ov; bestn = on; }
        }
        if (lane == 0) { s_val[wave] = best; s_idx[wave] = bestn; }
        __syncthreads();
        if (tid < 32) {
            float bv = s_val[tid]; int bn = s_idx[tid];
#pragma unroll
            for (int off = 16; off >= 1; off >>= 1) {
                float ov = __shfl_xor(bv, off, 32);
                int   on = __shfl_xor(bn, off, 32);
                if (ov > bv || (ov == bv && on < bn)) { bv = ov; bn = on; }
            }
            if (tid == 0) {
                fpsIdx[b * P + i] = bn;
                const float* p = xyz + ((size_t)b * N + bn) * 3;
                float x = p[0], y = p[1], z = p[2];
                newxyz[((size_t)b * P + i) * 3 + 0] = x;
                newxyz[((size_t)b * P + i) * 3 + 1] = y;
                newxyz[((size_t)b * P + i) * 3 + 2] = z;
                s_lastp[0] = x; s_lastp[1] = y; s_lastp[2] = z;
            }
        }
        __syncthreads();
    }
}

// ---------------------------------------------------------------------------
// Kernel 3: ball query, all three radii in one ascending scan per query point.
// Ascending scan == smallest-index nsample within radius (reference semantics).
// ---------------------------------------------------------------------------
__global__ void ballquery_kernel(const float* __restrict__ xyz,
                                 const float* __restrict__ newxyz,
                                 int* __restrict__ idx0, int* __restrict__ idx1,
                                 int* __restrict__ idx2) {
    const int N = 8192, BP = 4 * 2048;
    int q = blockIdx.x * blockDim.x + threadIdx.x;
    if (q >= BP) return;
    int b = q >> 11;
    float cx = newxyz[q * 3 + 0], cy = newxyz[q * 3 + 1], cz = newxyz[q * 3 + 2];
    const float r2a = 0.04f, r2b = 0.16f, r2c = 0.64f;
    int c0 = 0, c1 = 0, c2 = 0;
    int* o0 = idx0 + (size_t)q * 16;
    int* o1 = idx1 + (size_t)q * 32;
    int* o2 = idx2 + (size_t)q * 64;
    const float* X = xyz + (size_t)b * N * 3;
    for (int n = 0; n < N; ++n) {
        if ((n & 63) == 0) __builtin_prefetch(X + (n + 64) * 3, 0, 0);
        float dx = X[n * 3 + 0] - cx;
        float dy = X[n * 3 + 1] - cy;
        float dz = X[n * 3 + 2] - cz;
        float d2 = dx * dx + dy * dy + dz * dz;
        if (d2 < r2c) {
            if (c2 < 64) o2[c2++] = n;
            if (d2 < r2b && c1 < 32) o1[c1++] = n;
            if (d2 < r2a && c0 < 16) o0[c0++] = n;
            if (c0 == 16 && c1 == 32 && c2 == 64) break;
        }
    }
    int p0 = (c0 > 0) ? o0[0] : 0;
    int p1 = (c1 > 0) ? o1[0] : 0;
    int p2 = (c2 > 0) ? o2[0] : 0;
    for (int k = c0; k < 16; ++k) o0[k] = p0;
    for (int k = c1; k < 32; ++k) o1[k] = p1;
    for (int k = c2; k < 64; ++k) o2[k] = p2;
}

// ---------------------------------------------------------------------------
// Kernel 4: per-layer prep — pad+convert weights to bf16 (Cout x KPad),
// fold BN into per-channel scale/bias.
// ---------------------------------------------------------------------------
__global__ void prep_weights_kernel(const float* __restrict__ W,
                                    const float* __restrict__ gamma,
                                    const float* __restrict__ beta,
                                    const float* __restrict__ mean,
                                    const float* __restrict__ var,
                                    __bf16* __restrict__ Wb,
                                    float* __restrict__ bnsc, float* __restrict__ bnbi,
                                    int cout, int cin, int kpad) {
    int t = blockIdx.x * blockDim.x + threadIdx.x;
    if (t < cout * kpad) {
        int o = t / kpad, k = t % kpad;
        float v = (k < cin) ? W[o * cin + k] : 0.0f;
        Wb[t] = (__bf16)v;
    }
    if (t < cout) {
        float sc = gamma[t] * rsqrtf(var[t] + 1e-5f);
        bnsc[t] = sc;
        bnbi[t] = beta[t] - mean[t] * sc;
    }
}

// ---------------------------------------------------------------------------
// Fused group + 3-layer MLP (WMMA bf16, f32 accum) + max-pool.
// One 128-thread block (4 waves) per query point; n-tiles split across waves,
// the K-strip of B (weight) tiles held in registers across all m-tiles.
// ---------------------------------------------------------------------------
template <int KT>
static __device__ __forceinline__ void run_layer(
        const __bf16* inBuf, int ldin, __bf16* outBuf, int ldout,
        const __bf16* __restrict__ W, int mtiles, int ntiles,
        const float* __restrict__ bnsc, const float* __restrict__ bnbi,
        int wave, int lane) {
    constexpr int kpad = KT * 32;
    for (int nt = wave; nt < ntiles; nt += 4) {
        v16bf bw[KT];
#pragma unroll
        for (int kt = 0; kt < KT; ++kt) bw[kt] = load_b_tile(W, kpad, nt, kt, lane);
        float sc = bnsc[nt * 16 + (lane & 15)];
        float bi = bnbi[nt * 16 + (lane & 15)];
        int col = nt * 16 + (lane & 15);
        for (int mt = 0; mt < mtiles; ++mt) {
            v8f acc = {};
#pragma unroll
            for (int kt = 0; kt < KT; ++kt) {
                v16bf a = load_a_tile(inBuf, ldin, mt, kt, lane);
                acc = __builtin_amdgcn_wmma_f32_16x16x32_bf16(false, a, false, bw[kt],
                                                              (short)0, acc, false, false);
            }
            int mbase = mt * 16 + ((lane < 16) ? 0 : 8);
#pragma unroll
            for (int r = 0; r < 8; ++r) {
                float v = fmaxf(fmaf(acc[r], sc, bi), 0.0f);   // BN + ReLU
                outBuf[(mbase + r) * ldout + col] = (__bf16)v;
            }
        }
    }
}

template <int KT>
static __device__ __forceinline__ void run_layer_max(
        const __bf16* inBuf, int ldin,
        const __bf16* __restrict__ W, int mtiles, int ntiles,
        const float* __restrict__ bnsc, const float* __restrict__ bnbi,
        float* __restrict__ out, int b, int p, int chOff, int wave, int lane) {
    constexpr int kpad = KT * 32;
    for (int nt = wave; nt < ntiles; nt += 4) {
        v16bf bw[KT];
#pragma unroll
        for (int kt = 0; kt < KT; ++kt) bw[kt] = load_b_tile(W, kpad, nt, kt, lane);
        float sc = bnsc[nt * 16 + (lane & 15)];
        float bi = bnbi[nt * 16 + (lane & 15)];
        float vmax = 0.0f;                                  // ReLU floor
        for (int mt = 0; mt < mtiles; ++mt) {
            v8f acc = {};
#pragma unroll
            for (int kt = 0; kt < KT; ++kt) {
                v16bf a = load_a_tile(inBuf, ldin, mt, kt, lane);
                acc = __builtin_amdgcn_wmma_f32_16x16x32_bf16(false, a, false, bw[kt],
                                                              (short)0, acc, false, false);
            }
#pragma unroll
            for (int r = 0; r < 8; ++r)
                vmax = fmaxf(vmax, fmaf(acc[r], sc, bi));
        }
        // combine the two half-wave row groups (M 0..7 vs 8..15 of each tile)
        vmax = fmaxf(vmax, __shfl_xor(vmax, 16, 32));
        if (lane < 16)
            out[((size_t)b * 640 + chOff + nt * 16 + lane) * 2048 + p] = vmax;
    }
}

template <int S, int C1, int C2, int C3>
__global__ __launch_bounds__(128)
void fused_group_mlp_kernel(const float* __restrict__ xyz,
                            const float* __restrict__ newxyz,
                            const float* __restrict__ featT,
                            const int* __restrict__ ballIdx,
                            const __bf16* __restrict__ W1,
                            const __bf16* __restrict__ W2,
                            const __bf16* __restrict__ W3,
                            const float* __restrict__ bn,
                            float* __restrict__ out, int chOff) {
    constexpr int K1 = 96;                 // 67 padded to 96
    __shared__ alignas(32) __bf16 buf0[S * 128];
    __shared__ alignas(32) __bf16 buf1[S * 128];

    int q = blockIdx.x;                    // B*P == gridDim
    int b = q >> 11, p = q & 2047;
    int lane = threadIdx.x & 31, wave = threadIdx.x >> 5;

    float cx = newxyz[q * 3 + 0];
    float cy = newxyz[q * 3 + 1];
    float cz = newxyz[q * 3 + 2];
    const int* idx = ballIdx + (size_t)q * S;

    // gather x0: [rel xyz | 64 feats | zero pad] per neighbor, bf16, stride K1
    for (int e = threadIdx.x; e < S * K1; e += 128) {
        int s = e / K1, c = e % K1;
        int n = idx[s];
        float v;
        if (c < 3) {
            float ctr = (c == 0) ? cx : ((c == 1) ? cy : cz);
            v = xyz[((size_t)b * 8192 + n) * 3 + c] - ctr;
        } else if (c < 67) {
            v = featT[((size_t)b * 8192 + n) * 64 + (c - 3)];
        } else {
            v = 0.0f;
        }
        buf0[s * K1 + c] = (__bf16)v;
    }
    __syncthreads();

    const float* sc1 = bn;
    const float* bi1 = bn + C1;
    const float* sc2 = bn + 2 * C1;
    const float* bi2 = bn + 2 * C1 + C2;
    const float* sc3 = bn + 2 * C1 + 2 * C2;
    const float* bi3 = bn + 2 * C1 + 2 * C2 + C3;

    run_layer<K1 / 32>(buf0, K1, buf1, C1, W1, S / 16, C1 / 16, sc1, bi1, wave, lane);
    __syncthreads();
    run_layer<C1 / 32>(buf1, C1, buf0, C2, W2, S / 16, C2 / 16, sc2, bi2, wave, lane);
    __syncthreads();
    run_layer_max<C2 / 32>(buf0, C2, W3, S / 16, C3 / 16, sc3, bi3, out, b, p, chOff,
                           wave, lane);
}

// ---------------------------------------------------------------------------
// Host launcher
// ---------------------------------------------------------------------------
extern "C" void kernel_launch(void* const* d_in, const int* in_sizes, int n_in,
                              void* d_out, int out_size, void* d_ws, size_t ws_size,
                              hipStream_t stream) {
    const int B = 4, N = 8192, P = 2048;
    const float* xyz      = (const float*)d_in[0];
    const float* features = (const float*)d_in[1];

    // params: d_in[2 + (scale*3+layer)*5 + {W,gamma,beta,mean,var}]
    const float* Wp[9]; const float* Gp[9]; const float* Bp[9];
    const float* Mp[9]; const float* Vp[9];
    for (int L = 0; L < 9; ++L) {
        Wp[L] = (const float*)d_in[2 + L * 5 + 0];
        Gp[L] = (const float*)d_in[2 + L * 5 + 1];
        Bp[L] = (const float*)d_in[2 + L * 5 + 2];
        Mp[L] = (const float*)d_in[2 + L * 5 + 3];
        Vp[L] = (const float*)d_in[2 + L * 5 + 4];
    }
    static const int couts[9] = {64, 64, 128, 64, 128, 256, 128, 128, 256};
    static const int cins[9]  = {67, 64, 64,  67, 64,  128, 67,  128, 128};
    static const int kpads[9] = {96, 64, 64,  96, 64,  128, 96,  128, 128};

    // workspace carve-out
    uint8_t* ws = (uint8_t*)d_ws;
    size_t off = 0;
    auto take = [&](size_t bytes) -> void* {
        void* ptr = ws + off;
        off += (bytes + 255) & ~(size_t)255;
        return ptr;
    };
    float* featT  = (float*)take((size_t)B * N * 64 * sizeof(float));
    int*   fpsIdx = (int*)take((size_t)B * P * sizeof(int));
    int*   idx0   = (int*)take((size_t)B * P * 16 * sizeof(int));
    int*   idx1   = (int*)take((size_t)B * P * 32 * sizeof(int));
    int*   idx2   = (int*)take((size_t)B * P * 64 * sizeof(int));
    __bf16* Wb[9];
    for (int L = 0; L < 9; ++L)
        Wb[L] = (__bf16*)take((size_t)couts[L] * kpads[L] * sizeof(unsigned short));
    float* bnS[3];
    for (int s = 0; s < 3; ++s) {
        int tot = 2 * (couts[s * 3] + couts[s * 3 + 1] + couts[s * 3 + 2]);
        bnS[s] = (float*)take((size_t)tot * sizeof(float));
    }

    float* outNewXyz = (float*)d_out;                       // (B,P,3)
    float* outFeat   = (float*)d_out + (size_t)B * P * 3;   // (B,640,P)

    // 1) feature transpose
    {
        int total = B * 64 * N;
        transpose_feat_kernel<<<(total + 255) / 256, 256, 0, stream>>>(features, featT);
    }
    // 2) FPS (+new_xyz into d_out)
    fps_kernel<<<B, 1024, 0, stream>>>(xyz, fpsIdx, outNewXyz);
    // 3) ball query (all scales)
    ballquery_kernel<<<(B * P + 127) / 128, 128, 0, stream>>>(xyz, outNewXyz, idx0, idx1, idx2);
    // 4) weight prep + BN fold
    for (int s = 0; s < 3; ++s) {
        int c1 = couts[s * 3], c2 = couts[s * 3 + 1], c3 = couts[s * 3 + 2];
        float* scb[3][2] = {
            {bnS[s],                   bnS[s] + c1},
            {bnS[s] + 2 * c1,          bnS[s] + 2 * c1 + c2},
            {bnS[s] + 2 * c1 + 2 * c2, bnS[s] + 2 * c1 + 2 * c2 + c3}};
        for (int l = 0; l < 3; ++l) {
            int L = s * 3 + l;
            int total = couts[L] * kpads[L];
            prep_weights_kernel<<<(total + 255) / 256, 256, 0, stream>>>(
                Wp[L], Gp[L], Bp[L], Mp[L], Vp[L], Wb[L], scb[l][0], scb[l][1],
                couts[L], cins[L], kpads[L]);
        }
    }
    // 5) fused group + MLP + maxpool (WMMA), one block per query point
    fused_group_mlp_kernel<16, 64, 64, 128><<<B * P, 128, 0, stream>>>(
        xyz, outNewXyz, featT, idx0, Wb[0], Wb[1], Wb[2], bnS[0], outFeat, 0);
    fused_group_mlp_kernel<32, 64, 128, 256><<<B * P, 128, 0, stream>>>(
        xyz, outNewXyz, featT, idx1, Wb[3], Wb[4], Wb[5], bnS[1], outFeat, 128);
    fused_group_mlp_kernel<64, 128, 128, 256><<<B * P, 128, 0, stream>>>(
        xyz, outNewXyz, featT, idx2, Wb[6], Wb[7], Wb[8], bnS[2], outFeat, 384);
}
